// StudentMyHeCo_1657857376668
// MI455X (gfx1250) — matrix operand order repaired and checked by reading, not compiled, not run on hardware
//
#include <hip/hip_runtime.h>
#include <hip/hip_bf16.h>
#include <math.h>

typedef __attribute__((ext_vector_type(16))) _Float16 v16h;
typedef __attribute__((ext_vector_type(8)))  _Float16 v8h;
typedef __attribute__((ext_vector_type(8)))  float    v8f;
typedef __attribute__((ext_vector_type(4)))  float    v4f;

#define N_NODES 50000
#define F_DIM   1024
#define D       64
#define NVIEW   3
#define NEDGE   1600000
#define ROW_TILES (N_NODES / 16)   // 3125, exact
#define WAVES_PER_BLOCK 8
#define GEMM_BLOCKS ((ROW_TILES + WAVES_PER_BLOCK - 1) / WAVES_PER_BLOCK)

// fc-GEMM tiling
#define MT       4                       // row tiles (16 rows each) per wave
#define CHUNK_K  128                     // K-chunk staged in LDS
#define NCHUNK   (F_DIM / CHUNK_K)       // 8
#define BSM_PAD  8                       // halfs; row stride 272B -> conflict-free
#define FC_TILES_PER_BLOCK (WAVES_PER_BLOCK * MT)   // 32
#define FC_BLOCKS ((ROW_TILES + FC_TILES_PER_BLOCK - 1) / FC_TILES_PER_BLOCK)

// ---------------------------------------------------------------------------
// WMMA helpers (CDNA5 wave32, 16x16x32 f16 -> f32)
// ---------------------------------------------------------------------------
__device__ __forceinline__ v8f wmma_f16(v16h a, v16h b, v8f c) {
  // (neg_a, A, neg_b, B, c_mod, C, reuse_a, reuse_b)
  return __builtin_amdgcn_wmma_f32_16x16x32_f16(false, a, false, b, (short)0, c,
                                                false, false);
}

// B tile (32x16 f16): lane holds column n = lane&15, 8 K-halfs at +0 and 8 at
// +16 from k0 + (lane>>4)*8.  p = &W[d0 + (lane&15)][k0 + (lane>>4)*8]  (row-major W).
__device__ __forceinline__ v16h ldB16(const _Float16* __restrict__ p) {
  v8h lo = *(const v8h*)p;
  v8h hi = *(const v8h*)(p + 16);
  return __builtin_shufflevector(lo, hi, 0, 1, 2, 3, 4, 5, 6, 7,
                                 8, 9, 10, 11, 12, 13, 14, 15);
}

__device__ __forceinline__ v16h ldB_lds(const _Float16* p) {
  v8h lo = *(const v8h*)p;
  v8h hi = *(const v8h*)(p + 16);
  return __builtin_shufflevector(lo, hi, 0, 1, 2, 3, 4, 5, 6, 7,
                                 8, 9, 10, 11, 12, 13, 14, 15);
}

// A tile (16x32 f16) from f32 source: per lane, 8 floats at +0 and 8 at +16
// (K units); base already includes (lane>>4)*8.
__device__ __forceinline__ v16h ldA_f32(const float* __restrict__ p) {
  v4f a0 = *(const v4f*)(p);
  v4f a1 = *(const v4f*)(p + 4);
  v4f a2 = *(const v4f*)(p + 16);
  v4f a3 = *(const v4f*)(p + 20);
  v16h A;
#pragma unroll
  for (int i = 0; i < 4; ++i) {
    A[i]      = (_Float16)a0[i];
    A[4 + i]  = (_Float16)a1[i];
    A[8 + i]  = (_Float16)a2[i];
    A[12 + i] = (_Float16)a3[i];
  }
  return A;
}

__device__ __forceinline__ v16h ldA_f16(const _Float16* __restrict__ p) {
  v8h lo = *(const v8h*)(p);
  v8h hi = *(const v8h*)(p + 16);
  return __builtin_shufflevector(lo, hi, 0, 1, 2, 3, 4, 5, 6, 7,
                                 8, 9, 10, 11, 12, 13, 14, 15);
}

// ---------------------------------------------------------------------------
// f32 -> f16 weight conversion
// ---------------------------------------------------------------------------
__global__ void cvt_f16_kernel(const float* __restrict__ in,
                               _Float16* __restrict__ out, int n) {
  int i = blockIdx.x * blockDim.x + threadIdx.x;
  if (i < n) out[i] = (_Float16)in[i];
}

// ---------------------------------------------------------------------------
// h = elu(feats @ W0^T + b0), stored f16.
// One wave owns MT=4 16-row tiles (M=64); W0h staged through double-buffered
// LDS in 16KB K-chunks, so each block reads W0h from global exactly once and
// every B ds_load feeds 4 WMMAs.
// ---------------------------------------------------------------------------
__global__ __launch_bounds__(256, 1)
void fc_gemm_kernel(const float* __restrict__ feats,
                    const _Float16* __restrict__ w0h,
                    const float* __restrict__ b0,
                    _Float16* __restrict__ hout) {
  __shared__ __align__(16) _Float16 bsm[2][D][CHUNK_K + BSM_PAD];

  const int tid  = threadIdx.x;
  const int wave = tid >> 5;
  const int lane = tid & 31;
  const int m    = lane & 15;
  const int hi   = lane >> 4;

  // ---- stage chunk 0 of W0h into LDS (16KB: 1024 x 16B, 4 per thread) ----
  v8h stage[4];
#pragma unroll
  for (int j = 0; j < 4; ++j) {
    const int q = j * 256 + tid;                       // 0..1023
    stage[j] = *(const v8h*)(w0h + (size_t)(q >> 4) * F_DIM + (q & 15) * 8);
  }
#pragma unroll
  for (int j = 0; j < 4; ++j) {
    const int q = j * 256 + tid;
    *(v8h*)&bsm[0][q >> 4][(q & 15) * 8] = stage[j];
  }
  __syncthreads();

  // ---- per-wave row tiles (clamped; stores guarded) ----
  int tiles[MT];
  const float* arow[MT];
#pragma unroll
  for (int t = 0; t < MT; ++t) {
    tiles[t] = blockIdx.x * FC_TILES_PER_BLOCK + wave * MT + t;
    const int tt = tiles[t] < ROW_TILES ? tiles[t] : ROW_TILES - 1;
    arow[t] = feats + (size_t)(tt * 16 + m) * F_DIM + hi * 8;
  }
  v8f acc[MT][4] = {};

  for (int c = 0; c < NCHUNK; ++c) {
    const int buf = c & 1;
    // issue global loads for next chunk before computing (overlap)
    if (c + 1 < NCHUNK) {
#pragma unroll
      for (int j = 0; j < 4; ++j) {
        const int q = j * 256 + tid;
        stage[j] = *(const v8h*)(w0h + (size_t)(q >> 4) * F_DIM +
                                 (c + 1) * CHUNK_K + (q & 15) * 8);
      }
    }
#pragma unroll
    for (int t = 0; t < MT; ++t)
      __builtin_prefetch(arow[t] + (c + 1) * CHUNK_K, 0, 1);

#pragma unroll
    for (int ks = 0; ks < 4; ++ks) {
      const int kc = ks * 32;
      v16h A[MT];
#pragma unroll
      for (int t = 0; t < MT; ++t) A[t] = ldA_f32(arow[t] + c * CHUNK_K + kc);
#pragma unroll
      for (int g = 0; g < 4; ++g) {
        v16h B = ldB_lds(&bsm[buf][g * 16 + m][kc + hi * 8]);
#pragma unroll
        for (int t = 0; t < MT; ++t) acc[t][g] = wmma_f16(A[t], B, acc[t][g]);
      }
    }

    if (c + 1 < NCHUNK) {
      // buf^1 was last read in chunk c-1; the barrier at the end of that
      // iteration already separates those reads from these writes.
#pragma unroll
      for (int j = 0; j < 4; ++j) {
        const int q = j * 256 + tid;
        *(v8h*)&bsm[buf ^ 1][q >> 4][(q & 15) * 8] = stage[j];
      }
      __syncthreads();
    }
  }

  // ---- epilogue: bias + ELU, store f16 ----
#pragma unroll
  for (int t = 0; t < MT; ++t) {
    if (tiles[t] < ROW_TILES) {
      const int row0 = tiles[t] * 16;
#pragma unroll
      for (int g = 0; g < 4; ++g) {
        const int col = g * 16 + m;
        const float bias = b0[col];
#pragma unroll
        for (int r = 0; r < 8; ++r) {
          float v = acc[t][g][r] + bias;
          v = v > 0.0f ? v : expm1f(v);      // ELU
          hout[(size_t)(row0 + hi * 8 + r) * D + col] = (_Float16)v;
        }
      }
    }
  }
}

// ---------------------------------------------------------------------------
// s = h @ Wg[p]^T   (f16 in, f32 out), K = 64 (Wg is 8KB, L1-resident)
// ---------------------------------------------------------------------------
__global__ __launch_bounds__(256, 1)
void view_gemm_kernel(const _Float16* __restrict__ hin,
                      const _Float16* __restrict__ wgh,
                      float* __restrict__ s) {
  const int wave = threadIdx.x >> 5;
  const int lane = threadIdx.x & 31;
  const int tile = blockIdx.x * WAVES_PER_BLOCK + wave;
  if (tile >= ROW_TILES) return;
  const int row0 = tile * 16;
  const int m  = lane & 15;
  const int hi = lane >> 4;

  v8f acc[4] = {};
  const _Float16* arow  = hin + (size_t)(row0 + m) * D + hi * 8;
  const _Float16* bbase = wgh + (size_t)m * D + hi * 8;

#pragma unroll
  for (int k0 = 0; k0 < D; k0 += 32) {
    v16h A = ldA_f16(arow + k0);
#pragma unroll
    for (int g = 0; g < 4; ++g)
      acc[g] = wmma_f16(A, ldB16(bbase + k0 + (size_t)g * 16 * D), acc[g]);
  }

#pragma unroll
  for (int g = 0; g < 4; ++g) {
    const int col = g * 16 + m;
#pragma unroll
    for (int r = 0; r < 8; ++r)
      s[(size_t)(row0 + hi * 8 + r) * D + col] = acc[g][r];
  }
}

// ---------------------------------------------------------------------------
// SpMM: emb[rows[e]] += vals[e] * s[cols[e]].  16 threads per edge, float4
// gathers (L2-resident s), f32 global atomics (L2-resident emb).
// ---------------------------------------------------------------------------
__global__ __launch_bounds__(256, 1)
void spmm_kernel(const int* __restrict__ rows, const int* __restrict__ cols,
                 const float* __restrict__ vals, const float* __restrict__ s,
                 float* __restrict__ embp) {
  const long t = (long)blockIdx.x * blockDim.x + threadIdx.x;
  const int e = (int)(t >> 4);
  if (e >= NEDGE) return;
  const int g = ((int)t & 15) * 4;
  const int r = rows[e];
  const int c = cols[e];
  const float v = vals[e];
  v4f msg = *(const v4f*)(s + (size_t)c * D + g);
  float* o = embp + (size_t)r * D + g;
  atomicAdd(o + 0, v * msg.x);
  atomicAdd(o + 1, v * msg.y);
  atomicAdd(o + 2, v * msg.z);
  atomicAdd(o + 3, v * msg.w);
}

// ---------------------------------------------------------------------------
// emb[p] = prelu(emb[p] + bg[p], alpha[p])  over all P views
// ---------------------------------------------------------------------------
__global__ __launch_bounds__(256, 1)
void bias_prelu_kernel(float* __restrict__ emb, const float* __restrict__ bg,
                       const float* __restrict__ alpha) {
  const size_t tot = (size_t)NVIEW * N_NODES * D;
  size_t i = (size_t)blockIdx.x * blockDim.x + threadIdx.x;
  if (i >= tot) return;
  const int p = (int)(i / ((size_t)N_NODES * D));
  const int d = (int)(i & (D - 1));
  const float a = alpha[p];
  float v = emb[i] + bg[p * D + d];
  emb[i] = v > 0.0f ? v : a * v;
}

// ---------------------------------------------------------------------------
// spacc[p][d] += sum_n tanh((emb_p @ Wa^T)[n,d] + ba[d])   (WMMA + reduction)
// ---------------------------------------------------------------------------
__global__ __launch_bounds__(256, 1)
void att_gemm_kernel(const float* __restrict__ embp,
                     const _Float16* __restrict__ wah,
                     const float* __restrict__ ba,
                     float* __restrict__ spacc) {
  __shared__ float smem[D];
  const int tid = threadIdx.x;
  if (tid < D) smem[tid] = 0.0f;
  __syncthreads();

  const int wave = tid >> 5;
  const int lane = tid & 31;
  const int tile = blockIdx.x * WAVES_PER_BLOCK + wave;
  if (tile < ROW_TILES) {
    const int row0 = tile * 16;
    const int m  = lane & 15;
    const int hi = lane >> 4;

    v8f acc[4] = {};
    const float*    arow  = embp + (size_t)(row0 + m) * D + hi * 8;
    const _Float16* bbase = wah + (size_t)m * D + hi * 8;

#pragma unroll
    for (int k0 = 0; k0 < D; k0 += 32) {
      v16h A = ldA_f32(arow + k0);
#pragma unroll
      for (int g = 0; g < 4; ++g)
        acc[g] = wmma_f16(A, ldB16(bbase + k0 + (size_t)g * 16 * D), acc[g]);
    }

#pragma unroll
    for (int g = 0; g < 4; ++g) {
      const int col = g * 16 + m;
      const float bias = ba[col];
      float sum = 0.0f;
#pragma unroll
      for (int r = 0; r < 8; ++r) sum += tanhf(acc[g][r] + bias);
      atomicAdd(&smem[col], sum);          // ds_add_f32; lane & lane+16 same col
    }
  }
  __syncthreads();
  if (tid < D) atomicAdd(&spacc[tid], smem[tid]);
}

// ---------------------------------------------------------------------------
// beta = softmax_p( (spacc[p]/N) . att )   -- single wave
// ---------------------------------------------------------------------------
__global__ void beta_kernel(const float* __restrict__ spacc,
                            const float* __restrict__ att,
                            float* __restrict__ beta) {
  const int lane = threadIdx.x;   // 32 threads
  float sc[NVIEW];
#pragma unroll
  for (int p = 0; p < NVIEW; ++p) {
    float v = spacc[p * D + lane] * att[lane] +
              spacc[p * D + 32 + lane] * att[32 + lane];
#pragma unroll
    for (int off = 16; off > 0; off >>= 1) v += __shfl_down(v, off, 32);
    sc[p] = v;
  }
  if (lane == 0) {
    const float inv = 1.0f / (float)N_NODES;
    float s0 = sc[0] * inv, s1 = sc[1] * inv, s2 = sc[2] * inv;
    float mx = fmaxf(s0, fmaxf(s1, s2));
    float e0 = __expf(s0 - mx), e1 = __expf(s1 - mx), e2 = __expf(s2 - mx);
    float den = e0 + e1 + e2;
    beta[0] = e0 / den;
    beta[1] = e1 / den;
    beta[2] = e2 / den;
  }
}

// ---------------------------------------------------------------------------
// out = sum_p beta[p] * emb[p]   (float4 streaming)
// ---------------------------------------------------------------------------
__global__ __launch_bounds__(256, 1)
void combine_kernel(const float* __restrict__ emb,
                    const float* __restrict__ beta,
                    float* __restrict__ out) {
  const size_t n4 = (size_t)N_NODES * D / 4;
  size_t i = (size_t)blockIdx.x * blockDim.x + threadIdx.x;
  if (i >= n4) return;
  const float b0 = beta[0], b1 = beta[1], b2 = beta[2];
  v4f v0 = ((const v4f*)emb)[i];
  v4f v1 = ((const v4f*)emb)[i + n4];
  v4f v2 = ((const v4f*)emb)[i + 2 * n4];
  v4f o;
  o.x = b0 * v0.x + b1 * v1.x + b2 * v2.x;
  o.y = b0 * v0.y + b1 * v1.y + b2 * v2.y;
  o.z = b0 * v0.z + b1 * v1.z + b2 * v2.z;
  o.w = b0 * v0.w + b1 * v1.w + b2 * v2.w;
  ((v4f*)out)[i] = o;
}

// ---------------------------------------------------------------------------
// host launcher
// ---------------------------------------------------------------------------
extern "C" void kernel_launch(void* const* d_in, const int* in_sizes, int n_in,
                              void* d_out, int out_size, void* d_ws,
                              size_t ws_size, hipStream_t stream) {
  const float* feats = (const float*)d_in[0];
  const int*   rows  = (const int*)d_in[1];
  const int*   cols  = (const int*)d_in[2];
  const float* vals  = (const float*)d_in[3];
  const float* W0    = (const float*)d_in[4];
  const float* b0    = (const float*)d_in[5];
  const float* Wg    = (const float*)d_in[6];
  const float* bg    = (const float*)d_in[7];
  const float* alpha = (const float*)d_in[8];
  const float* Wa    = (const float*)d_in[9];
  const float* ba    = (const float*)d_in[10];
  const float* att   = (const float*)d_in[11];

  char* ws = (char*)d_ws;
  size_t cur = 0;
  auto carve = [&](size_t bytes) -> void* {
    cur = (cur + 255) & ~(size_t)255;
    void* p = ws + cur;
    cur += bytes;
    return p;
  };

  _Float16* w0h   = (_Float16*)carve((size_t)D * F_DIM * 2);
  _Float16* wgh   = (_Float16*)carve((size_t)NVIEW * D * D * 2);
  _Float16* wah   = (_Float16*)carve((size_t)D * D * 2);
  float*    spacc = (float*)carve((size_t)NVIEW * D * 4);
  float*    beta  = (float*)carve(4 * 4);
  _Float16* hbuf  = (_Float16*)carve((size_t)N_NODES * D * 2);
  float*    sbuf  = (float*)carve((size_t)N_NODES * D * 4);
  float*    emb   = (float*)carve((size_t)NVIEW * N_NODES * D * 4);

  hipMemsetAsync(emb, 0, (size_t)NVIEW * N_NODES * D * 4, stream);
  hipMemsetAsync(spacc, 0, (size_t)NVIEW * D * 4, stream);

  cvt_f16_kernel<<<(D * F_DIM + 255) / 256, 256, 0, stream>>>(W0, w0h, D * F_DIM);
  cvt_f16_kernel<<<(NVIEW * D * D + 255) / 256, 256, 0, stream>>>(Wg, wgh,
                                                                  NVIEW * D * D);
  cvt_f16_kernel<<<(D * D + 255) / 256, 256, 0, stream>>>(Wa, wah, D * D);

  fc_gemm_kernel<<<FC_BLOCKS, 256, 0, stream>>>(feats, w0h, b0, hbuf);

  for (int p = 0; p < NVIEW; ++p) {
    view_gemm_kernel<<<GEMM_BLOCKS, 256, 0, stream>>>(
        hbuf, wgh + (size_t)p * D * D, sbuf);
    const long tspmm = (long)NEDGE * 16;
    spmm_kernel<<<(int)((tspmm + 255) / 256), 256, 0, stream>>>(
        rows + (size_t)p * NEDGE, cols + (size_t)p * NEDGE,
        vals + (size_t)p * NEDGE, sbuf, emb + (size_t)p * N_NODES * D);
  }

  {
    const size_t tot = (size_t)NVIEW * N_NODES * D;
    bias_prelu_kernel<<<(int)((tot + 255) / 256), 256, 0, stream>>>(emb, bg,
                                                                    alpha);
  }

  for (int p = 0; p < NVIEW; ++p)
    att_gemm_kernel<<<GEMM_BLOCKS, 256, 0, stream>>>(
        emb + (size_t)p * N_NODES * D, wah, ba, spacc + p * D);

  beta_kernel<<<1, 32, 0, stream>>>(spacc, att, beta);

  {
    const size_t n4 = (size_t)N_NODES * D / 4;
    combine_kernel<<<(int)((n4 + 255) / 256), 256, 0, stream>>>(emb, beta,
                                                                (float*)d_out);
  }
}
